// GraphSAGE_33320356282736
// MI455X (gfx1250) — matrix operand order repaired and checked by reading, not compiled, not run on hardware
//
#include <hip/hip_runtime.h>

// ---------------------------------------------------------------------------
// GraphSAGE (3x SAGEConv mean-aggr + relu + dropout) for MI455X / gfx1250.
// Aggregation: f32 global atomics (HBM/L2 bound). Dense parts: bf16 WMMA
// (v_wmma_f32_16x16x32_bf16), f32 accumulate, fused bias/relu/dropout epilogue.
// ---------------------------------------------------------------------------

#define NNODES 50000
#define NEDGES 800000
#define D_IN   128
#define D_H    128
#define D_OUT  64

typedef __attribute__((ext_vector_type(4)))  __bf16 v4bf;
typedef __attribute__((ext_vector_type(8)))  __bf16 v8bf;
typedef __attribute__((ext_vector_type(16))) __bf16 v16bf;
typedef __attribute__((ext_vector_type(8)))  float  v8f;
typedef __attribute__((ext_vector_type(4)))  float  v4f;

// ---------------- small utility kernels ----------------

__global__ void k_zero_f32(float* __restrict__ p, int n4) {
    int i = blockIdx.x * blockDim.x + threadIdx.x;
    if (i < n4) { v4f z = {}; *(v4f*)(p + 4 * (size_t)i) = z; }
}

__global__ void k_count(const int* __restrict__ dst, float* __restrict__ cnt) {
    int e = blockIdx.x * blockDim.x + threadIdx.x;
    if (e < NEDGES) atomicAdd(&cnt[dst[e]], 1.0f);
}

__global__ void k_recip(float* __restrict__ rc) {
    int n = blockIdx.x * blockDim.x + threadIdx.x;
    if (n < NNODES) rc[n] = 1.0f / fmaxf(rc[n], 1.0f);
}

// f32 (N x 128) -> bf16
__global__ void k_cast_bf16(const float* __restrict__ in, __bf16* __restrict__ out, int n4) {
    int i = blockIdx.x * blockDim.x + threadIdx.x;
    if (i >= n4) return;
    v4f v = *(const v4f*)(in + 4 * (size_t)i);
    v4bf o;
    o[0] = (__bf16)v[0]; o[1] = (__bf16)v[1]; o[2] = (__bf16)v[2]; o[3] = (__bf16)v[3];
    *(v4bf*)(out + 4 * (size_t)i) = o;
}

// convert one layer's weights wl, wr (each dout*din f32, row-major [out][in]) to bf16
__global__ void k_cast_w(const float* __restrict__ wl, const float* __restrict__ wr,
                         __bf16* __restrict__ wbf, int sz) {
    int i = blockIdx.x * blockDim.x + threadIdx.x;
    if (i < sz) {
        wbf[i]      = (__bf16)wl[i];
        wbf[sz + i] = (__bf16)wr[i];
    }
}

// scatter-add of bf16 features into f32 sums; one wave per edge, 4 feats/lane
__global__ void k_scatter(const __bf16* __restrict__ h, const int* __restrict__ src,
                          const int* __restrict__ dst, float* __restrict__ sum) {
    long long t = blockIdx.x * (long long)blockDim.x + threadIdx.x;
    if (t >= (long long)NEDGES * 32) return;
    int e = (int)(t >> 5);
    int c = ((int)t & 31) * 4;          // feature chunk 0..124
    int s = src[e], d = dst[e];
    v4bf v = *(const v4bf*)(h + (size_t)s * D_IN + c);
    float* sp = sum + (size_t)d * D_IN + c;
    atomicAdd(sp + 0, (float)v[0]);
    atomicAdd(sp + 1, (float)v[1]);
    atomicAdd(sp + 2, (float)v[2]);
    atomicAdd(sp + 3, (float)v[3]);
}

// mean = sum * rc[node], cast to bf16
__global__ void k_meancast(const float* __restrict__ sum, const float* __restrict__ rc,
                           __bf16* __restrict__ mb, int n4) {
    int i = blockIdx.x * blockDim.x + threadIdx.x;
    if (i >= n4) return;
    int node = (i * 4) >> 7;            // 128 feats per node
    float r = rc[node];
    v4f v = *(const v4f*)(sum + 4 * (size_t)i);
    v4bf o;
    o[0] = (__bf16)(v[0] * r); o[1] = (__bf16)(v[1] * r);
    o[2] = (__bf16)(v[2] * r); o[3] = (__bf16)(v[3] * r);
    *(v4bf*)(mb + 4 * (size_t)i) = o;
}

// ---------------- WMMA GEMM: out = mean @ wl.T + h @ wr.T + b (+relu+dropout) ----
// One wave per 16x16 output tile. blockDim = (32, 4): 4 column-tiles per block.
// MODE 0: hidden layer -> relu, dropout(u >= 0.3)/0.7, write bf16.
// MODE 1: final layer  -> write f32.

template <int DIN, int DOUT, int MODE>
__global__ __launch_bounds__(128)
void k_gemm(const __bf16* __restrict__ hin, const __bf16* __restrict__ meanb,
            const __bf16* __restrict__ wbf,      // [2][DOUT][DIN] bf16 (wl then wr)
            const float* __restrict__ bias,
            const float* __restrict__ u,         // dropout uniforms (N x DOUT) or nullptr
            __bf16* __restrict__ hout,           // MODE 0 output
            float* __restrict__ fout)            // MODE 1 output
{
    const int lane = threadIdx.x;                // 0..31
    const int m0   = blockIdx.x * 16;            // node-row tile (N divisible by 16)
    const int j0   = (blockIdx.y * blockDim.y + threadIdx.y) * 16;
    const int nn   = lane & 15;
    const int hi   = lane >> 4;

    const __bf16* wl = wbf;
    const __bf16* wr = wbf + (size_t)DOUT * DIN;

    // A fragment: lane holds row m0+nn, 16 contiguous K starting at koff = hi*16
    const int koff = hi * 16;
    const __bf16* aM = meanb + (size_t)(m0 + nn) * DIN;
    const __bf16* aH = hin   + (size_t)(m0 + nn) * DIN;
    // B fragment: lane holds column j0+nn, 16 contiguous K starting at hi*16
    const __bf16* bL = wl + (size_t)(j0 + nn) * DIN + hi * 16;
    const __bf16* bR = wr + (size_t)(j0 + nn) * DIN + hi * 16;

    v8f acc = {};
#pragma unroll
    for (int kc = 0; kc < DIN; kc += 32) {
        v8bf mlo = *(const v8bf*)(aM + kc + koff);
        v8bf mhi = *(const v8bf*)(aM + kc + koff + 8);
        v16bf aFragM = __builtin_shufflevector(mlo, mhi,
                         0,1,2,3,4,5,6,7,8,9,10,11,12,13,14,15);
        v16bf bFragL = *(const v16bf*)(bL + kc);
        acc = __builtin_amdgcn_wmma_f32_16x16x32_bf16(
                  false, aFragM, false, bFragL, (short)0, acc, false, false);

        v8bf hlo = *(const v8bf*)(aH + kc + koff);
        v8bf hhi = *(const v8bf*)(aH + kc + koff + 8);
        v16bf aFragH = __builtin_shufflevector(hlo, hhi,
                         0,1,2,3,4,5,6,7,8,9,10,11,12,13,14,15);
        v16bf bFragR = *(const v16bf*)(bR + kc);
        acc = __builtin_amdgcn_wmma_f32_16x16x32_bf16(
                  false, aFragH, false, bFragR, (short)0, acc, false, false);
    }

    // Epilogue. C layout: lane column = nn, VGPR r -> row = 8*hi + r.
    const float b = bias[j0 + nn];
    const int col = j0 + nn;
#pragma unroll
    for (int r = 0; r < 8; ++r) {
        int row = m0 + hi * 8 + r;
        float v = acc[r] + b;
        if (MODE == 0) {
            v = v > 0.0f ? v : 0.0f;                         // relu
            float uu = u[(size_t)row * DOUT + col];          // dropout (training)
            v = (uu >= 0.3f) ? v * (1.0f / 0.7f) : 0.0f;
            hout[(size_t)row * DOUT + col] = (__bf16)v;
        } else {
            fout[(size_t)row * DOUT + col] = v;
        }
    }
}

// ---------------- host-side orchestration ----------------

extern "C" void kernel_launch(void* const* d_in, const int* in_sizes, int n_in,
                              void* d_out, int out_size, void* d_ws, size_t ws_size,
                              hipStream_t stream) {
    (void)in_sizes; (void)n_in; (void)out_size; (void)ws_size;

    const float* x   = (const float*)d_in[0];
    const float* u1  = (const float*)d_in[1];
    const float* u2  = (const float*)d_in[2];
    const float* wl0 = (const float*)d_in[3];
    const float* bl0 = (const float*)d_in[4];
    const float* wr0 = (const float*)d_in[5];
    const float* wl1 = (const float*)d_in[6];
    const float* bl1 = (const float*)d_in[7];
    const float* wr1 = (const float*)d_in[8];
    const float* wl2 = (const float*)d_in[9];
    const float* bl2 = (const float*)d_in[10];
    const float* wr2 = (const float*)d_in[11];
    const int*   ei  = (const int*)d_in[12];
    const int* src = ei;
    const int* dst = ei + NEDGES;
    float* out = (float*)d_out;

    // workspace carve-up
    const size_t F = (size_t)NNODES * 128;
    float*  rc    = (float*)d_ws;                 // N
    float*  sum   = rc + NNODES;                  // N*128 f32
    __bf16* hA    = (__bf16*)(sum + F);           // N*128 bf16
    __bf16* hB    = hA + F;                       // N*128 bf16
    __bf16* meanb = hB + F;                       // N*128 bf16
    __bf16* wbf   = meanb + F;                    // 2*128*128 bf16

    const int T = 256;
    const int n4   = (int)(F / 4);                // 1.6M
    const int gF4  = (n4 + T - 1) / T;

    // ---- once: degrees -> reciprocal, cast x -> bf16 ----
    k_zero_f32<<<(NNODES / 4 + T - 1) / T, T, 0, stream>>>(rc, NNODES / 4);
    k_count<<<(NEDGES + T - 1) / T, T, 0, stream>>>(dst, rc);
    k_recip<<<(NNODES + T - 1) / T, T, 0, stream>>>(rc);
    k_cast_bf16<<<gF4, T, 0, stream>>>(x, hA, n4);

    const long long scT = (long long)NEDGES * 32;
    const int scG = (int)((scT + T - 1) / T);

    // ---- layer 0: hA -> hB ----
    k_zero_f32<<<gF4, T, 0, stream>>>(sum, n4);
    k_scatter<<<scG, T, 0, stream>>>(hA, src, dst, sum);
    k_meancast<<<gF4, T, 0, stream>>>(sum, rc, meanb, n4);
    k_cast_w<<<(D_H * D_IN + T - 1) / T, T, 0, stream>>>(wl0, wr0, wbf, D_H * D_IN);
    k_gemm<128, 128, 0><<<dim3(NNODES / 16, 2), dim3(32, 4), 0, stream>>>(
        hA, meanb, wbf, bl0, u1, hB, nullptr);

    // ---- layer 1: hB -> hA ----
    k_zero_f32<<<gF4, T, 0, stream>>>(sum, n4);
    k_scatter<<<scG, T, 0, stream>>>(hB, src, dst, sum);
    k_meancast<<<gF4, T, 0, stream>>>(sum, rc, meanb, n4);
    k_cast_w<<<(D_H * D_H + T - 1) / T, T, 0, stream>>>(wl1, wr1, wbf, D_H * D_H);
    k_gemm<128, 128, 0><<<dim3(NNODES / 16, 2), dim3(32, 4), 0, stream>>>(
        hB, meanb, wbf, bl1, u2, hA, nullptr);

    // ---- layer 2: hA -> out (f32, N x 64) ----
    k_zero_f32<<<gF4, T, 0, stream>>>(sum, n4);
    k_scatter<<<scG, T, 0, stream>>>(hA, src, dst, sum);
    k_meancast<<<gF4, T, 0, stream>>>(sum, rc, meanb, n4);
    k_cast_w<<<(D_OUT * D_H + T - 1) / T, T, 0, stream>>>(wl2, wr2, wbf, D_OUT * D_H);
    k_gemm<128, 64, 1><<<dim3(NNODES / 16, 1), dim3(32, 4), 0, stream>>>(
        hA, meanb, wbf, bl2, nullptr, nullptr, out);
}
